// CustomGATConv_87531433492503
// MI455X (gfx1250) — matrix operand order repaired and checked by reading, not compiled, not run on hardware
//
#include <hip/hip_runtime.h>
#include <hip/hip_bf16.h>

typedef __attribute__((ext_vector_type(16))) _Float16 v16h;
typedef __attribute__((ext_vector_type(8)))  _Float16 v8h;
typedef __attribute__((ext_vector_type(8)))  float    v8f;

#define N_NODES 8192
#define IN_DIM  200
#define OUT_DIM 64
#define NEG_INF_F (-1.0e9f)

// ---------------------------------------------------------------------------
// Kernel 1: H = X@W + b  -> Hh (f16 row-major [N,64]) and HhT (f16 [64,N]).
// ---------------------------------------------------------------------------
__global__ __launch_bounds__(64) void gat_proj_kernel(
    const float* __restrict__ X, const float* __restrict__ W,
    const float* __restrict__ b, _Float16* __restrict__ Hh,
    _Float16* __restrict__ HhT) {
  const int row = blockIdx.x;
  const int j   = threadIdx.x;            // output feature 0..63
  float acc = b[j];
  const float* xr = X + (size_t)row * IN_DIM;
#pragma unroll 8
  for (int k = 0; k < IN_DIM; ++k)
    acc = fmaf(xr[k], W[(size_t)k * OUT_DIM + j], acc);
  _Float16 h = (_Float16)acc;
  Hh [(size_t)row * OUT_DIM + j] = h;
  HhT[(size_t)j * N_NODES + row] = h;
}

static __device__ inline v8f wmma_f16(v16h a, v16h b, v8f c) {
  // D = A(16x32 f16) * B(32x16 f16) + C(16x16 f32)
  return __builtin_amdgcn_wmma_f32_16x16x32_f16(false, a, false, b,
                                                (short)0, c, false, false);
}

// ---------------------------------------------------------------------------
// Kernel 2: fused masked attention, two-pass flash.
// 1 workgroup = 16 rows, 4 waves, each wave owns 2048 columns, 32 cols/iter.
// Pass 1: scores via WMMA (pre-scaled by log2e through the A-fragment),
//         stream A once (NT), lane-local row max, mask bits in lane-private
//         registers (bit position == iteration).
// Pass 2: recompute scores (Hh is L2-hot), p = exp2(s' - m') via bare
//         v_exp_f32, lane-local l-sum, P via LDS re-layout, 4x WMMA P*V.
// No per-iteration cross-lane ops, no accumulator rescaling.
// ---------------------------------------------------------------------------
__global__ __launch_bounds__(128) void gat_attn_kernel(
    const int* __restrict__ A, const _Float16* __restrict__ Hh,
    const _Float16* __restrict__ HhT, float* __restrict__ out) {
  __shared__ _Float16 ptile[4][16][32];     // per-wave P transpose buffer
  __shared__ float red_m[4][32][8];
  __shared__ float red_l[4][32][8];
  __shared__ float red_acc[4][32][32];      // [wave][lane][t*8+g]

  const int w    = threadIdx.x >> 5;
  const int lane = threadIdx.x & 31;
  const int half = lane >> 4;               // 0: rows g, 1: rows g+8
  const int l16  = lane & 15;
  const int rowBase = blockIdx.x * 16;
  const int kb     = half * 8;              // unused; A-frags also use khoff
  const int khoff  = half * 16;             // B-frag K sub-offset per half-wave

  // Resident A-matrix fragments for this block's 16 rows (K = 0..63),
  // pre-scaled by log2(e) so softmax can use exp2 (saves a v_mul per exp).
  const _Float16* hrow = Hh + (size_t)(rowBase + l16) * OUT_DIM;
  v8h alo0 = *(const v8h*)(hrow + khoff);
  v8h alo1 = *(const v8h*)(hrow + khoff + 8);
  v8h ahi0 = *(const v8h*)(hrow + khoff + 32);
  v8h ahi1 = *(const v8h*)(hrow + khoff + 40);
  v8h sc8;
#pragma unroll
  for (int i = 0; i < 8; ++i) sc8[i] = (_Float16)1.44269504f;
  alo0 *= sc8; alo1 *= sc8; ahi0 *= sc8; ahi1 *= sc8;
  v16h a_lo = __builtin_shufflevector(alo0, alo1, 0,1,2,3,4,5,6,7,8,9,10,11,12,13,14,15);
  v16h a_hi = __builtin_shufflevector(ahi0, ahi1, 0,1,2,3,4,5,6,7,8,9,10,11,12,13,14,15);

  const int colStart = w * (N_NODES / 4);
  const int* Abase = A + (size_t)(rowBase + half * 8) * N_NODES + l16;

  // Lane-private mask bits: [chunk][g][tile], bit position == it (0..31).
  unsigned mw[2][8][2];
  float vmax[8];
#pragma unroll
  for (int g = 0; g < 8; ++g) {
    vmax[g] = -INFINITY;
    mw[0][g][0] = 0u; mw[0][g][1] = 0u;
    mw[1][g][0] = 0u; mw[1][g][1] = 0u;
  }

  // ------------------------- pass 1: masks + row max -------------------------
#pragma unroll
  for (int ch = 0; ch < 2; ++ch) {
    for (int it = 0; it < 32; ++it) {
      const int c = colStart + ch * 1024 + it * 32;
      __builtin_prefetch(Abase + (size_t)c + 32, 0, 0);

      v16h b00 = *(const v16h*)(Hh + (size_t)(c      + l16) * OUT_DIM + khoff);
      v16h b01 = *(const v16h*)(Hh + (size_t)(c      + l16) * OUT_DIM + 32 + khoff);
      v16h b10 = *(const v16h*)(Hh + (size_t)(c + 16 + l16) * OUT_DIM + khoff);
      v16h b11 = *(const v16h*)(Hh + (size_t)(c + 16 + l16) * OUT_DIM + 32 + khoff);

      v8f s0 = {}, s1 = {};
      s0 = wmma_f16(a_lo, b00, s0);
      s0 = wmma_f16(a_hi, b01, s0);
      s1 = wmma_f16(a_lo, b10, s1);
      s1 = wmma_f16(a_hi, b11, s1);

#pragma unroll
      for (int g = 0; g < 8; ++g) {
        const int* ar = Abase + (size_t)g * N_NODES + c;
        int a0 = __builtin_nontemporal_load(ar);       // values are 0/1
        int a1 = __builtin_nontemporal_load(ar + 16);
        mw[ch][g][0] += (unsigned)a0 << it;            // v_lshl_add
        mw[ch][g][1] += (unsigned)a1 << it;
        float v0 = (a0 > 0) ? s0[g] : NEG_INF_F;
        float v1 = (a1 > 0) ? s1[g] : NEG_INF_F;
        vmax[g] = fmaxf(vmax[g], fmaxf(v0, v1));
      }
    }
  }

  // One butterfly per row (rows live in 16-lane groups), then cross-wave max.
#pragma unroll
  for (int g = 0; g < 8; ++g) {
    float v = vmax[g];
#pragma unroll
    for (int off = 8; off >= 1; off >>= 1)
      v = fmaxf(v, __shfl_xor(v, off, 32));
    red_m[w][lane][g] = v;
  }
  __syncthreads();
  float m[8];
#pragma unroll
  for (int g = 0; g < 8; ++g) {
    float v = fmaxf(red_m[0][lane][g], red_m[1][lane][g]);
    v = fmaxf(v, fmaxf(red_m[2][lane][g], red_m[3][lane][g]));
    m[g] = v;                               // global row max (log2-scaled)
  }

  // ------------------------- pass 2: exp2 + P*V ------------------------------
  v8f acc0 = {}, acc1 = {}, acc2 = {}, acc3 = {};
  float lsum[8];
#pragma unroll
  for (int g = 0; g < 8; ++g) lsum[g] = 0.f;

#pragma unroll
  for (int ch = 0; ch < 2; ++ch) {
    for (int it = 0; it < 32; ++it) {
      const int c = colStart + ch * 1024 + it * 32;

      v16h b00 = *(const v16h*)(Hh + (size_t)(c      + l16) * OUT_DIM + khoff);
      v16h b01 = *(const v16h*)(Hh + (size_t)(c      + l16) * OUT_DIM + 32 + khoff);
      v16h b10 = *(const v16h*)(Hh + (size_t)(c + 16 + l16) * OUT_DIM + khoff);
      v16h b11 = *(const v16h*)(Hh + (size_t)(c + 16 + l16) * OUT_DIM + 32 + khoff);

      v8f s0 = {}, s1 = {};
      s0 = wmma_f16(a_lo, b00, s0);
      s0 = wmma_f16(a_hi, b01, s0);
      s1 = wmma_f16(a_lo, b10, s1);
      s1 = wmma_f16(a_hi, b11, s1);

#pragma unroll
      for (int g = 0; g < 8; ++g) {
        const bool b0 = (mw[ch][g][0] >> it) & 1u;
        const bool b1 = (mw[ch][g][1] >> it) & 1u;
        float v0 = b0 ? s0[g] : NEG_INF_F;
        float v1 = b1 ? s1[g] : NEG_INF_F;
        float p0 = __builtin_amdgcn_exp2f(v0 - m[g]);  // == exp(s - m), <= 1
        float p1 = __builtin_amdgcn_exp2f(v1 - m[g]);
        lsum[g] += p0 + p1;
        const int pr = g + 8 * half;
        ptile[w][pr][l16]      = (_Float16)p0;
        ptile[w][pr][16 + l16] = (_Float16)p1;
      }

      // Re-read P as a 16x32 A-matrix fragment (D->A re-layout via LDS).
      const _Float16* prow = &ptile[w][l16][0];
      v8h pl0 = *(const v8h*)(prow + khoff);
      v8h pl1 = *(const v8h*)(prow + khoff + 8);
      v16h pf = __builtin_shufflevector(pl0, pl1, 0,1,2,3,4,5,6,7,8,9,10,11,12,13,14,15);

      // V B-fragments from transposed H: B(k,n) = HhT[t*16+n][c+k].
      v16h bv0 = *(const v16h*)(HhT + (size_t)( 0 + l16) * N_NODES + c + khoff);
      v16h bv1 = *(const v16h*)(HhT + (size_t)(16 + l16) * N_NODES + c + khoff);
      v16h bv2 = *(const v16h*)(HhT + (size_t)(32 + l16) * N_NODES + c + khoff);
      v16h bv3 = *(const v16h*)(HhT + (size_t)(48 + l16) * N_NODES + c + khoff);

      acc0 = wmma_f16(pf, bv0, acc0);
      acc1 = wmma_f16(pf, bv1, acc1);
      acc2 = wmma_f16(pf, bv2, acc2);
      acc3 = wmma_f16(pf, bv3, acc3);
    }
  }

  // One l-sum butterfly per row; cross-wave combine is a plain sum.
#pragma unroll
  for (int g = 0; g < 8; ++g) {
    float v = lsum[g];
#pragma unroll
    for (int off = 8; off >= 1; off >>= 1)
      v += __shfl_xor(v, off, 32);
    red_l[w][lane][g] = v;
    red_acc[w][lane][0 * 8 + g] = acc0[g];
    red_acc[w][lane][1 * 8 + g] = acc1[g];
    red_acc[w][lane][2 * 8 + g] = acc2[g];
    red_acc[w][lane][3 * 8 + g] = acc3[g];
  }
  __syncthreads();

  const int t = w;                          // wave w finalizes feature tile w
#pragma unroll
  for (int g = 0; g < 8; ++g) {
    float lt = (red_l[0][lane][g] + red_l[1][lane][g]) +
               (red_l[2][lane][g] + red_l[3][lane][g]);
    float av = (red_acc[0][lane][t * 8 + g] + red_acc[1][lane][t * 8 + g]) +
               (red_acc[2][lane][t * 8 + g] + red_acc[3][lane][t * 8 + g]);
    float o = av / lt;
    o = (o > 0.f) ? o : 0.f;                // ReLU
    out[(size_t)(rowBase + g + 8 * half) * OUT_DIM + t * 16 + l16] = o;
  }
}

extern "C" void kernel_launch(void* const* d_in, const int* in_sizes, int n_in,
                              void* d_out, int out_size, void* d_ws, size_t ws_size,
                              hipStream_t stream) {
  const float* X = (const float*)d_in[0];   // [8192, 200] f32
  const int*   A = (const int*)d_in[1];     // [8192, 8192] i32
  const float* W = (const float*)d_in[2];   // [200, 64] f32
  const float* b = (const float*)d_in[3];   // [64] f32
  float* out = (float*)d_out;               // [8192, 64] f32

  _Float16* Hh  = (_Float16*)d_ws;                       // 1 MB
  _Float16* HhT = Hh + (size_t)N_NODES * OUT_DIM;        // 1 MB

  gat_proj_kernel<<<N_NODES, 64, 0, stream>>>(X, W, b, Hh, HhT);
  gat_attn_kernel<<<N_NODES / 16, 128, 0, stream>>>(A, Hh, HhT, out);
}
